// ShakespeareNet_53919019434253
// MI455X (gfx1250) — matrix operand order repaired and checked by reading, not compiled, not run on hardware
//
#include <hip/hip_runtime.h>

// ---------------- CDNA5 WMMA types ----------------
typedef __attribute__((ext_vector_type(16))) __bf16 bf16x16;
typedef __attribute__((ext_vector_type(8)))  __bf16 bf16x8;
typedef __attribute__((ext_vector_type(8)))  float  f32x8;

#define WMMA_BF16(a, b, c) \
  __builtin_amdgcn_wmma_f32_16x16x32_bf16(false, (a), false, (b), (short)0, (c), false, false)

// ---------------- problem sizes ----------------
#define HDIM   256
#define G4     1024
#define TSTEPS 80
#define BTOT   2048
#define VOC    90
#define VPAD   96
#define BM     32      // batch rows per workgroup
#define NWAVES 16      // 512 threads
#define PITCH  264     // bf16 elems per LDS row (256 + 8 pad), 528B = 33*16B
#define HBUF   (BM * PITCH)

__device__ __forceinline__ f32x8 splat8(float v) {
  f32x8 r;
#pragma unroll
  for (int i = 0; i < 8; ++i) r[i] = v;
  return r;
}

__device__ __forceinline__ bf16x16 cat8(bf16x8 lo, bf16x8 hi) {
  return __builtin_shufflevector(lo, hi, 0,1,2,3,4,5,6,7,8,9,10,11,12,13,14,15);
}

// A-matrix 16x32 bf16 layout (ISA 7.12.2): lane m=l%16, half=l/16,
// elems 0..7 = K[kb + 8*half .. +7], elems 8..15 = K[kb + 16 + 8*half .. +7]
__device__ __forceinline__ bf16x16 loadA(const __bf16* base, int m, int kb, int half) {
  const __bf16* p = base + m * PITCH + kb + half * 8;
  return cat8(*(const bf16x8*)p, *(const bf16x8*)(p + 16));
}

// B-matrix 32x16 bf16: lane n=l%16, half=l/16, elems 0..7 = K[kb + 8*half .. +7],
// elems 8..15 = K[kb + 16 + 8*half .. +7]; source W is row-major (n, k) with k-stride 256
__device__ __forceinline__ bf16x16 loadB(const __bf16* W, int n, int kb, int half) {
  const __bf16* p = W + n * 256 + kb + half * 8;
  return cat8(*(const bf16x8*)p, *(const bf16x8*)(p + 16));
}

// One K=256 GEMM pass: acc[mt][gate] += A(LDS h) x B(global bf16 W)
// kb loop kept rolled (unroll 1): bounds live B-fragments to one clause of 4,
// avoiding the VGPR blowup + scratch spills seen with full unroll.
__device__ __forceinline__ void gemm_pass(f32x8 (&acc)[2][4], const __bf16* aBuf,
                                          const __bf16* W, int l16, int half, int jbase) {
#pragma unroll 1
  for (int kb = 0; kb < 256; kb += 32) {
    bf16x16 a0 = loadA(aBuf, l16,      kb, half);
    bf16x16 a1 = loadA(aBuf, 16 + l16, kb, half);
#pragma unroll
    for (int g = 0; g < 4; ++g) {
      bf16x16 b = loadB(W, g * 256 + jbase + l16, kb, half);
      acc[0][g] = WMMA_BF16(a0, b, acc[0][g]);
      acc[1][g] = WMMA_BF16(a1, b, acc[1][g]);
    }
  }
}

// ---- branch-free transcendentals: CDNA5 has V_TANH_F32 (TRANS unit, co-executes
// ---- with XDL/WMMA). Sigmoid via tanh: sigma(x) = 0.5*tanh(x/2) + 0.5.
#if __has_builtin(__builtin_amdgcn_tanhf)
__device__ __forceinline__ float fast_tanh(float x) { return __builtin_amdgcn_tanhf(x); }
#else
__device__ __forceinline__ float fast_tanh(float x) { return tanhf(x); }
#endif
__device__ __forceinline__ float sigm(float x) { return __builtin_fmaf(fast_tanh(0.5f * x), 0.5f, 0.5f); }

// LSTM pointwise: gates (i,f,g,o) in acc -> h, c_new  (C/D layout: elem e -> row half*8+e)
__device__ __forceinline__ void cell_update(const f32x8 (&acc)[2][4], const f32x8 (&cOld)[2],
                                            f32x8 (&h)[2], f32x8 (&cN)[2]) {
#pragma unroll
  for (int mt = 0; mt < 2; ++mt) {
#pragma unroll
    for (int e = 0; e < 8; ++e) {
      float iv = sigm(acc[mt][0][e]);
      float fv = sigm(acc[mt][1][e]);
      float gv = fast_tanh(acc[mt][2][e]);
      float ov = sigm(acc[mt][3][e]);
      float cn = fv * cOld[mt][e] + iv * gv;
      cN[mt][e] = cn;
      h[mt][e]  = ov * fast_tanh(cn);
    }
  }
}

// ---------------- persistent scan kernel ----------------
// LDS h buffers: 0 = lh1, 1 = gh1(mixed), 2 = lh2, 3 = gh2(mixed)
__global__ __launch_bounds__(512, 1) void lstm_persistent(
    const int* __restrict__ seq,
    const float* __restrict__ XWg, const float* __restrict__ XWl,
    const float* __restrict__ b2g, const float* __restrict__ b2l,
    const float* __restrict__ plA, const float* __restrict__ pgA,
    const __bf16* __restrict__ Wg1, const __bf16* __restrict__ Wl1,
    const __bf16* __restrict__ Wig2, const __bf16* __restrict__ Whg2,
    const __bf16* __restrict__ Wil2, const __bf16* __restrict__ Whl2,
    const __bf16* __restrict__ fcWp, const float* __restrict__ fcb,
    float* __restrict__ out) {
  __shared__ __bf16 sH[4 * HBUF];
  __shared__ float  sPl[BM];
  __shared__ float  sPg[BM];

  const int tid   = threadIdx.x;
  const int wave  = tid >> 5;
  const int lane  = tid & 31;
  const int half  = lane >> 4;
  const int l16   = lane & 15;
  const int jbase = wave * 16;                 // hidden slice owned by this wave
  const int rowb  = blockIdx.x * BM;           // batch base

  // zero h state; stage mixing probabilities in LDS (keeps them out of VGPRs)
  for (int i = tid; i < 4 * HBUF; i += 512) sH[i] = (__bf16)0.0f;
  if (tid < BM) { sPl[tid] = plA[rowb + tid]; sPg[tid] = pgA[rowb + tid]; }

  float b2gr[4], b2lr[4];
#pragma unroll
  for (int g = 0; g < 4; ++g) {
    b2gr[g] = b2g[g * 256 + jbase + l16];
    b2lr[g] = b2l[g * 256 + jbase + l16];
  }
  const int fc_mt = wave / 6, fc_nt = wave % 6;
  const int fc_v  = fc_nt * 16 + l16;
  const float fcbv = (wave < 12 && fc_v < VOC) ? fcb[fc_v] : 0.0f;

  // c states (registers, C/D layout, wave's 16-wide j slice)
  f32x8 cG1[2], cL1[2], cG2[2], cL2[2];
#pragma unroll
  for (int mt = 0; mt < 2; ++mt) {
    cG1[mt] = splat8(0.f); cL1[mt] = splat8(0.f);
    cG2[mt] = splat8(0.f); cL2[mt] = splat8(0.f);
  }
  __syncthreads();

#pragma unroll 1
  for (int t = 0; t < TSTEPS; ++t) {
    // ---- tokens for this step (scattered, L2-hot) ----
    int tok[2][8];
#pragma unroll
    for (int mt = 0; mt < 2; ++mt)
#pragma unroll
      for (int e = 0; e < 8; ++e)
        tok[mt][e] = seq[(rowb + mt * 16 + half * 8 + e) * TSTEPS + t];

    f32x8 acc[2][4], hG[2], cGn[2], hL[2], cLn[2];

    // ---- layer 1, cell g1: init from token table (incl. biases), add h @ Whh ----
#pragma unroll
    for (int mt = 0; mt < 2; ++mt)
#pragma unroll
      for (int g = 0; g < 4; ++g)
#pragma unroll
        for (int e = 0; e < 8; ++e)
          acc[mt][g][e] = XWg[tok[mt][e] * G4 + (g * 256 + jbase + l16)];
    gemm_pass(acc, sH + 1 * HBUF, Wg1, l16, half, jbase);
    cell_update(acc, cG1, hG, cGn);

    // ---- layer 1, cell l1 ----
#pragma unroll
    for (int mt = 0; mt < 2; ++mt)
#pragma unroll
      for (int g = 0; g < 4; ++g)
#pragma unroll
        for (int e = 0; e < 8; ++e)
          acc[mt][g][e] = XWl[tok[mt][e] * G4 + (g * 256 + jbase + l16)];
    gemm_pass(acc, sH + 0 * HBUF, Wl1, l16, half, jbase);
    cell_update(acc, cL1, hL, cLn);

    __syncthreads();  // all layer-1 LDS reads complete
#pragma unroll
    for (int mt = 0; mt < 2; ++mt)
#pragma unroll
      for (int e = 0; e < 8; ++e) {
        int r = mt * 16 + half * 8 + e, c = jbase + l16;
        float pl = sPl[r], pg = sPg[r];
        float hm = pl * hL[mt][e] + pg * hG[mt][e];
        cG1[mt][e] = pl * cLn[mt][e] + pg * cGn[mt][e];
        cL1[mt][e] = cLn[mt][e];
        sH[0 * HBUF + r * PITCH + c] = (__bf16)hL[mt][e];
        sH[1 * HBUF + r * PITCH + c] = (__bf16)hm;
      }
    __syncthreads();  // new h1 visible

    // ---- layer 2, cell g2: gh1_mix @ Wih + gh2 @ Whh + bias ----
#pragma unroll
    for (int mt = 0; mt < 2; ++mt)
#pragma unroll
      for (int g = 0; g < 4; ++g) acc[mt][g] = splat8(b2gr[g]);
    gemm_pass(acc, sH + 1 * HBUF, Wig2, l16, half, jbase);
    gemm_pass(acc, sH + 3 * HBUF, Whg2, l16, half, jbase);
    cell_update(acc, cG2, hG, cGn);

    // ---- layer 2, cell l2 ----
#pragma unroll
    for (int mt = 0; mt < 2; ++mt)
#pragma unroll
      for (int g = 0; g < 4; ++g) acc[mt][g] = splat8(b2lr[g]);
    gemm_pass(acc, sH + 0 * HBUF, Wil2, l16, half, jbase);
    gemm_pass(acc, sH + 2 * HBUF, Whl2, l16, half, jbase);
    cell_update(acc, cL2, hL, cLn);

    __syncthreads();  // all layer-2 LDS reads complete
#pragma unroll
    for (int mt = 0; mt < 2; ++mt)
#pragma unroll
      for (int e = 0; e < 8; ++e) {
        int r = mt * 16 + half * 8 + e, c = jbase + l16;
        float pl = sPl[r], pg = sPg[r];
        float hm = pl * hL[mt][e] + pg * hG[mt][e];
        cG2[mt][e] = pl * cLn[mt][e] + pg * cGn[mt][e];
        cL2[mt][e] = cLn[mt][e];
        sH[2 * HBUF + r * PITCH + c] = (__bf16)hL[mt][e];
        sH[3 * HBUF + r * PITCH + c] = (__bf16)hm;
      }
    __syncthreads();  // new h2 (= hidden output) visible

    // ---- fused FC: out[:, v, t] = gh2_mix @ fcW.T + fcb  (12 tiles / 16 waves) ----
    if (wave < 12) {
      f32x8 facc = splat8(fcbv);
#pragma unroll 1
      for (int kb = 0; kb < 256; kb += 32) {
        bf16x16 a = loadA(sH + 3 * HBUF, fc_mt * 16 + l16, kb, half);
        bf16x16 b = loadB(fcWp, fc_nt * 16 + l16, kb, half);
        facc = WMMA_BF16(a, b, facc);
      }
      if (fc_v < VOC) {
#pragma unroll
        for (int e = 0; e < 8; ++e) {
          int r = rowb + fc_mt * 16 + half * 8 + e;
          out[r * (VOC * TSTEPS) + fc_v * TSTEPS + t] = facc[e];
        }
      }
    }
  }
}

// ---------------- prep kernels ----------------
__global__ void prep_mlp(const int* __restrict__ cid, const float* __restrict__ cemb,
                         const float* pW0, const float* pb0, const float* pW1, const float* pb1,
                         const float* pW2, const float* pb2, const float* pW3, const float* pb3,
                         float* __restrict__ pl, float* __restrict__ pg,
                         float* __restrict__ out2) {
  int b = blockIdx.x * blockDim.x + threadIdx.x;
  if (b >= BTOT) return;
  int c = cid[b];
  float x0[10];
  for (int i = 0; i < 10; ++i) x0[i] = cemb[c * 10 + i];
  float h0[100];
  for (int o = 0; o < 100; ++o) {
    float s = pb0[o];
    for (int i = 0; i < 10; ++i) s += pW0[o * 10 + i] * x0[i];
    h0[o] = fmaxf(s, 0.f);
  }
  float h1[50];
  for (int o = 0; o < 50; ++o) {
    float s = pb1[o];
    for (int i = 0; i < 100; ++i) s += pW1[o * 100 + i] * h0[i];
    h1[o] = fmaxf(s, 0.f);
  }
  float h2[25];
  for (int o = 0; o < 25; ++o) {
    float s = pb2[o];
    for (int i = 0; i < 50; ++i) s += pW2[o * 50 + i] * h1[i];
    h2[o] = fmaxf(s, 0.f);
  }
  float l0 = pb3[0], l1 = pb3[1];
  for (int i = 0; i < 25; ++i) { l0 += pW3[i] * h2[i]; l1 += pW3[25 + i] * h2[i]; }
  float m = fmaxf(l0, l1);
  float e0 = __expf(l0 - m), e1 = __expf(l1 - m);
  float inv = 1.f / (e0 + e1);
  float plv = e0 * inv, pgv = e1 * inv;
  pl[b] = plv; pg[b] = pgv;
  for (int t = 0; t < TSTEPS; ++t) {
    out2[b * (2 * TSTEPS) + 2 * t]     = plv;
    out2[b * (2 * TSTEPS) + 2 * t + 1] = pgv;
  }
}

__global__ void cvt_bf16(const float* __restrict__ s, __bf16* __restrict__ d, int n) {
  int i = blockIdx.x * blockDim.x + threadIdx.x;
  if (i < n) d[i] = (__bf16)s[i];
}

__global__ void cvt_fc(const float* __restrict__ fcW, __bf16* __restrict__ d) {
  int i = blockIdx.x * blockDim.x + threadIdx.x;
  if (i >= VPAD * 256) return;
  int v = i >> 8;
  d[i] = (v < VOC) ? (__bf16)fcW[i] : (__bf16)0.0f;
}

// token tables: XW[v][n] = emb[v] . Wih[n] + bih[n] + bhh[n]
__global__ void build_tab(const float* __restrict__ emb,
                          const float* WihG, const float* bihG, const float* bhhG,
                          const float* WihL, const float* bihL, const float* bhhL,
                          float* __restrict__ XWg, float* __restrict__ XWl) {
  int id = blockIdx.x * blockDim.x + threadIdx.x;
  if (id >= 2 * VOC * G4) return;
  int c = id / (VOC * G4);
  int r = id - c * (VOC * G4);
  int v = r >> 10, n = r & 1023;
  const float* W = c ? WihL : WihG;
  float s = c ? (bihL[n] + bhhL[n]) : (bihG[n] + bhhG[n]);
  for (int k = 0; k < 8; ++k) s += emb[v * 8 + k] * W[n * 8 + k];
  (c ? XWl : XWg)[v * G4 + n] = s;
}

__global__ void build_b2(const float* bihG, const float* bhhG,
                         const float* bihL, const float* bhhL,
                         float* __restrict__ b2g, float* __restrict__ b2l) {
  int i = blockIdx.x * blockDim.x + threadIdx.x;
  if (i < G4) { b2g[i] = bihG[i] + bhhG[i]; b2l[i] = bihL[i] + bhhL[i]; }
}

// ---------------- launcher ----------------
extern "C" void kernel_launch(void* const* d_in, const int* in_sizes, int n_in,
                              void* d_out, int out_size, void* d_ws, size_t ws_size,
                              hipStream_t stream) {
  (void)in_sizes; (void)n_in; (void)out_size; (void)ws_size;
  const int*   seq  = (const int*)d_in[0];
  const int*   cid  = (const int*)d_in[1];
  const float* emb  = (const float*)d_in[2];
  const float* cemb = (const float*)d_in[3];
  const float* Wih_g1 = (const float*)d_in[4];
  const float* Whh_g1 = (const float*)d_in[5];
  const float* bih_g1 = (const float*)d_in[6];
  const float* bhh_g1 = (const float*)d_in[7];
  const float* Wih_g2 = (const float*)d_in[8];
  const float* Whh_g2 = (const float*)d_in[9];
  const float* bih_g2 = (const float*)d_in[10];
  const float* bhh_g2 = (const float*)d_in[11];
  const float* Wih_l1 = (const float*)d_in[12];
  const float* Whh_l1 = (const float*)d_in[13];
  const float* bih_l1 = (const float*)d_in[14];
  const float* bhh_l1 = (const float*)d_in[15];
  const float* Wih_l2 = (const float*)d_in[16];
  const float* Whh_l2 = (const float*)d_in[17];
  const float* bih_l2 = (const float*)d_in[18];
  const float* bhh_l2 = (const float*)d_in[19];
  const float* pW0 = (const float*)d_in[20]; const float* pb0 = (const float*)d_in[21];
  const float* pW1 = (const float*)d_in[22]; const float* pb1 = (const float*)d_in[23];
  const float* pW2 = (const float*)d_in[24]; const float* pb2 = (const float*)d_in[25];
  const float* pW3 = (const float*)d_in[26]; const float* pb3 = (const float*)d_in[27];
  const float* fcW = (const float*)d_in[28]; const float* fcb = (const float*)d_in[29];

  char* ws = (char*)d_ws;
  float* pl   = (float*)(ws + 0);
  float* pg   = (float*)(ws + 8192);
  float* XWg  = (float*)(ws + 16384);
  float* XWl  = (float*)(ws + 385024);
  float* b2g  = (float*)(ws + 753664);
  float* b2l  = (float*)(ws + 757760);
  __bf16* Wg1b  = (__bf16*)(ws + 761856);
  __bf16* Wl1b  = Wg1b  + 262144;
  __bf16* Wig2b = Wl1b  + 262144;
  __bf16* Whg2b = Wig2b + 262144;
  __bf16* Wil2b = Whg2b + 262144;
  __bf16* Whl2b = Wil2b + 262144;
  __bf16* fcWp  = Whl2b + 262144;

  float* out1 = (float*)d_out;
  float* out2 = out1 + (size_t)BTOT * VOC * TSTEPS;

  prep_mlp<<<8, 256, 0, stream>>>(cid, cemb, pW0, pb0, pW1, pb1, pW2, pb2, pW3, pb3,
                                  pl, pg, out2);
  cvt_bf16<<<1024, 256, 0, stream>>>(Whh_g1, Wg1b, 262144);
  cvt_bf16<<<1024, 256, 0, stream>>>(Whh_l1, Wl1b, 262144);
  cvt_bf16<<<1024, 256, 0, stream>>>(Wih_g2, Wig2b, 262144);
  cvt_bf16<<<1024, 256, 0, stream>>>(Whh_g2, Whg2b, 262144);
  cvt_bf16<<<1024, 256, 0, stream>>>(Wih_l2, Wil2b, 262144);
  cvt_bf16<<<1024, 256, 0, stream>>>(Whh_l2, Whl2b, 262144);
  cvt_fc<<<96, 256, 0, stream>>>(fcW, fcWp);
  build_tab<<<(2 * VOC * G4 + 255) / 256, 256, 0, stream>>>(
      emb, Wih_g1, bih_g1, bhh_g1, Wih_l1, bih_l1, bhh_l1, XWg, XWl);
  build_b2<<<4, 256, 0, stream>>>(bih_g2, bhh_g2, bih_l2, bhh_l2, b2g, b2l);

  lstm_persistent<<<BTOT / BM, 512, 0, stream>>>(
      seq, XWg, XWl, b2g, b2l, pl, pg,
      Wg1b, Wl1b, Wig2b, Whg2b, Wil2b, Whl2b, fcWp, fcb, out1);
}
